// Ultra45KCNN_QNN_Model_22033182228576
// MI455X (gfx1250) — compile-verified, matrix operand order checked by reading, not run on hardware
//
#include <hip/hip_runtime.h>
#include <hip/hip_fp16.h>
#include <math.h>

// ---------------------------------------------------------------------------
// Ultra45K CNN + "quantum" QNN, fused for gfx1250 (MI455X).
//
// The quantum circuit (7 layers RY + CNOT chain) is a fixed linear operator U
// (256x256):  qout = sum_{n<128}(U s)_n^2 - sum_{n>=128}(U s)_n^2.
// All dense layers run on v_wmma_f32_16x16x32_f16 with B-matrices pre-packed
// into WMMA fragment order (f16). Packed U is staged into LDS once per block
// via the Tensor Data Mover (tensor_load_to_lds + s_wait_tensorcnt).
// ---------------------------------------------------------------------------

typedef __attribute__((ext_vector_type(16))) _Float16     v16h;
typedef __attribute__((ext_vector_type(8)))  float        v8f;
typedef __attribute__((ext_vector_type(4)))  unsigned int v4u;
typedef __attribute__((ext_vector_type(4)))  int          v4i;
typedef __attribute__((ext_vector_type(8)))  int          v8i;

#define TPB 256
#define SPB 128   // samples per block (8 waves x 16 M-rows)

// -------------------- LDS layout (bytes) -----------------------------------
#define OFF_WB  0
#define OFF_R1  3648
#define OFF_R2  (OFF_R1 + 150528)
#define OFF_R3  (OFF_R2 + 75264)
#define OFF_QS  (OFF_R3 + 57344)
#define SMEM_BYTES (OFF_QS + 512)            // 287296 B < 320KB WGP LDS
// During qsim, R2+R3 (132608 B, contiguous) hold packed U (131072 B).

#define N_QUBITS 8
#define N_LAYERS 7
#define QDIM 256

// -------------------- workspace layout (halves) ----------------------------
#define WS_UPK 0
#define WS_FCP 65536
#define WS_C2P (WS_FCP + 16384)
#define WS_C3P (WS_C2P + 27648)
#define WS_C4P (WS_C3P + 17920)

#define HAVE_TDM __has_builtin(__builtin_amdgcn_tensor_load_to_lds)

// ===========================================================================
// Kernel 1: build U (256x256) from q_weights by evolving the identity in LDS,
// then emit it directly in WMMA B-fragment order (f16) to workspace.
// ===========================================================================
__global__ __launch_bounds__(TPB, 1)
void build_u_kernel(const float* __restrict__ qw, _Float16* __restrict__ Upk) {
    extern __shared__ float S[];       // [256][256]  row = amplitude, col = basis
    const int t = threadIdx.x;

    for (int idx = t; idx < QDIM * QDIM; idx += TPB)
        S[idx] = ((idx >> 8) == (idx & 255)) ? 1.0f : 0.0f;
    __syncthreads();

    for (int layer = 0; layer < N_LAYERS; ++layer) {
        // RY on qubit q (bit 7-q): pair stride 2^(7-q)
        for (int q = 0; q < N_QUBITS; ++q) {
            const float th = 0.5f * qw[layer * N_QUBITS + q];
            const float c = cosf(th), s = sinf(th);
            const int half = 1 << (7 - q);
            for (int p = t; p < 128 * QDIM; p += TPB) {
                const int col = p & 255;
                const int pr  = p >> 8;
                const int a   = pr / half, cc = pr % half;
                const int i0  = a * (half << 1) + cc;
                const int i1  = i0 + half;
                const float s0 = S[i0 * QDIM + col];
                const float s1 = S[i1 * QDIM + col];
                S[i0 * QDIM + col] = c * s0 - s * s1;
                S[i1 * QDIM + col] = s * s0 + c * s1;
            }
            __syncthreads();
        }
        // CNOT(q, q+1): swap (b1=1,b2=0) <-> (b1=1,b2=1)
        for (int q = 0; q < N_QUBITS - 1; ++q) {
            const int h = 1 << (6 - q);
            for (int p = t; p < 64 * QDIM; p += TPB) {
                const int col = p & 255;
                const int pr  = p >> 8;
                const int a   = pr / h, cc = pr % h;
                const int i0  = a * (h << 2) + (h << 1) + cc;
                const int i1  = i0 + h;
                const float tmp = S[i0 * QDIM + col];
                S[i0 * QDIM + col] = S[i1 * QDIM + col];
                S[i1 * QDIM + col] = tmp;
            }
            __syncthreads();
        }
    }
    // pack U into fragment order: idx = ((nt*8 + ks)*32 + lane)*16 + i
    for (int idx = t; idx < 65536; idx += TPB) {
        const int i    = idx & 15;
        const int lane = (idx >> 4) & 31;
        const int ks   = (idx >> 9) & 7;
        const int nt   = idx >> 12;
        const int half = lane >> 4, l15 = lane & 15;
        const int ncol = nt * 16 + l15;
        const int k    = ks * 32 + half * 8 + (i < 8 ? i : i + 8);
        Upk[idx] = (_Float16)S[ncol * QDIM + k];
    }
}

// ===========================================================================
// Pack a row-major (N x K) f32 weight matrix into WMMA B-fragment f16 order.
// ===========================================================================
__global__ void pack_w_kernel(const float* __restrict__ W, int N, int K,
                              _Float16* __restrict__ dst) {
    const int ntiles = (N + 15) >> 4;
    const int ksteps = (K + 31) >> 5;
    const int total  = ntiles * ksteps * 512;
    for (int idx = blockIdx.x * blockDim.x + threadIdx.x; idx < total;
         idx += gridDim.x * blockDim.x) {
        const int i    = idx & 15;
        const int lane = (idx >> 4) & 31;
        const int rest = idx >> 9;
        const int ks   = rest % ksteps;
        const int nt   = rest / ksteps;
        const int half = lane >> 4, l15 = lane & 15;
        const int ncol = nt * 16 + l15;
        const int k    = ks * 32 + half * 8 + (i < 8 ? i : i + 8);
        dst[idx] = (ncol < N && k < K) ? (_Float16)W[ncol * K + k] : (_Float16)0.0f;
    }
}

// ===========================================================================
// TDM: DMA 131072 bytes (packed U) from global into LDS at byte offset
// lds_off. 2D descriptor: 1 row of 32768 4-byte elements.
// ===========================================================================
#if HAVE_TDM
__device__ __forceinline__
void tdm_load_upk(const _Float16* __restrict__ gsrc, unsigned lds_off) {
    const unsigned long long ga = (unsigned long long)(const void*)gsrc;
    v4u g0;
    g0[0] = 1u;                                        // count=1, user D#
    g0[1] = lds_off;                                   // lds_addr (bytes)
    g0[2] = (unsigned)(ga & 0xffffffffu);              // global_addr[31:0]
    g0[3] = (unsigned)((ga >> 32) & 0x01ffffffu)       // global_addr[56:32]
            | (2u << 30);                              // type = 2 ("image")
    v8i g1;
    g1[0] = (2 << 16);                                 // data_size=4B, mask=0
    g1[1] = (int)0x80000000u;                          // tensor_dim0[15:0]=32768
    g1[2] = (1 << 16);                                 // dim0 hi=0, tensor_dim1=1
    g1[3] = (int)0x80000000u;                          // tile_dim0 = 32768
    g1[4] = 1;                                         // tile_dim1=1, tile_dim2=0
    g1[5] = 32768;                                     // tensor_dim0_stride lo
    g1[6] = (int)0x80000000u;                          // stride0 hi=0, stride1 lo
    g1[7] = 0;                                         // stride1 hi
    const v4i gz = {0, 0, 0, 0};
#if defined(__clang_major__) && (__clang_major__ >= 23)
    const v8i gz8 = {0, 0, 0, 0, 0, 0, 0, 0};
    __builtin_amdgcn_tensor_load_to_lds(g0, g1, gz, gz, gz8, 0);
#else
    __builtin_amdgcn_tensor_load_to_lds(g0, g1, gz, gz, 0);
#endif
}
#endif

// ===========================================================================
// Templated WMMA GEMM: out[m][n] = relu(bias[n] + sum_k A[m][k]*W[n][k]).
// ===========================================================================
template<int K, int N>
__device__ __forceinline__
void gemm_bias_relu(const float* __restrict__ A_lds, int lda,
                    const _Float16* __restrict__ Wp,
                    const float* __restrict__ bias,
                    float* __restrict__ out_lds, int ldo,
                    int lane, int m0) {
    constexpr int NT = (N + 15) / 16;
    constexpr int KS = (K + 31) / 32;
    const int half = lane >> 4;
    const int l15  = lane & 15;
    const float* arow = A_lds + (m0 + l15) * lda;

    v16h afr[KS];
#pragma unroll
    for (int ks = 0; ks < KS; ++ks) {
        const int kbase = ks * 32 + half * 8;
#pragma unroll
        for (int i = 0; i < 8; ++i) {
            const int k1 = kbase + i, k2 = k1 + 16;
            afr[ks][i]     = (k1 < K) ? (_Float16)arow[k1] : (_Float16)0.0f;
            afr[ks][8 + i] = (k2 < K) ? (_Float16)arow[k2] : (_Float16)0.0f;
        }
    }
#pragma unroll
    for (int nt = 0; nt < NT; ++nt) {
        v8f acc = {};
#pragma unroll
        for (int ks = 0; ks < KS; ++ks) {
            const v16h b = *(const v16h*)(Wp + ((nt * KS + ks) * 32 + lane) * 16);
            acc = __builtin_amdgcn_wmma_f32_16x16x32_f16(
                false, afr[ks], false, b, (short)0, acc, false, false);
        }
        const int nc = nt * 16 + l15;
        if (nc < N) {
            const float bv = bias[nc];
#pragma unroll
            for (int r = 0; r < 8; ++r)
                out_lds[(m0 + half * 8 + r) * ldo + nc] = fmaxf(acc[r] + bv, 0.0f);
        }
    }
}

// qsim: y = U s (WMMA, B fragments from LDS), qacc[m] += sign(n)*y^2,
// reduced across lanes via ds_add_f32.
__device__ __forceinline__
void qsim_accum(const float* __restrict__ A_lds,          // feat [128][256]
                const _Float16* __restrict__ Ul,          // packed U in LDS
                float* __restrict__ qout_lds,             // [128]
                int lane, int m0) {
    const int half = lane >> 4;
    const int l15  = lane & 15;
    const float* arow = A_lds + (m0 + l15) * QDIM;

    v16h afr[8];
#pragma unroll
    for (int ks = 0; ks < 8; ++ks) {
        const int kbase = ks * 32 + half * 8;
#pragma unroll
        for (int i = 0; i < 8; ++i) {
            afr[ks][i]     = (_Float16)arow[kbase + i];
            afr[ks][8 + i] = (_Float16)arow[kbase + 16 + i];
        }
    }
    float qacc[8] = {0.f, 0.f, 0.f, 0.f, 0.f, 0.f, 0.f, 0.f};
    for (int nt = 0; nt < 16; ++nt) {
        const float sign = ((nt * 16 + l15) < 128) ? 1.0f : -1.0f;
        v8f acc = {};
#pragma unroll
        for (int ks = 0; ks < 8; ++ks) {
            const v16h b = *(const v16h*)(Ul + ((nt * 8 + ks) * 32 + lane) * 16);
            acc = __builtin_amdgcn_wmma_f32_16x16x32_f16(
                false, afr[ks], false, b, (short)0, acc, false, false);
        }
#pragma unroll
        for (int r = 0; r < 8; ++r) qacc[r] += sign * acc[r] * acc[r];
    }
#pragma unroll
    for (int r = 0; r < 8; ++r)
        atomicAdd(&qout_lds[m0 + half * 8 + r], qacc[r]);   // ds_add_f32
}

// ===========================================================================
// Kernel 2: fully fused pipeline, 128 samples per 256-thread block.
// ===========================================================================
__global__ __launch_bounds__(TPB, 1)
void fused_kernel(const float* __restrict__ x,
                  const float* __restrict__ c1w, const float* __restrict__ c1b,
                  const float* __restrict__ c2w, const float* __restrict__ c2b,
                  const float* __restrict__ c3w, const float* __restrict__ c3b,
                  const _Float16* __restrict__ fcp, const float* __restrict__ fcb,
                  const _Float16* __restrict__ Upk,
                  const float* __restrict__ w1c, const float* __restrict__ b1c,
                  const _Float16* __restrict__ c2p, const float* __restrict__ b2c,
                  const _Float16* __restrict__ c3p, const float* __restrict__ b3c,
                  const _Float16* __restrict__ c4p, const float* __restrict__ b4c,
                  const float* __restrict__ w5c, const float* __restrict__ b5c,
                  float* __restrict__ out) {
    extern __shared__ char smem[];
    float*    WB  = (float*)(smem + OFF_WB);
    _Float16* R1h = (_Float16*)(smem + OFF_R1);
    float*    R1f = (float*)(smem + OFF_R1);
    _Float16* R2h = (_Float16*)(smem + OFF_R2);
    float*    R2f = (float*)(smem + OFF_R2);
    float*    R3f = (float*)(smem + OFF_R3);
    float*    qsh = (float*)(smem + OFF_QS);
    _Float16* Ul  = (_Float16*)(smem + OFF_R2);   // packed U staged in R2+R3

    const int t    = threadIdx.x;
    const int lane = t & 31;
    const int m0   = (t >> 5) * 16;
    const int s0   = blockIdx.x * SPB;

    // ---- prefetch the block's input stream (global_prefetch_b8) ----------
    {
        const char* xb = (const char*)(x + (size_t)s0 * 784);
        const int bytes = SPB * 784 * 4;
        for (int off = t * 256; off < bytes; off += TPB * 256)
            __builtin_prefetch(xb + off, 0, 3);
    }

    // ---- stage small conv params into LDS --------------------------------
    {
        const float* srcs[6] = {c1w, c1b, c2w, c2b, c3w, c3b};
        const int    cnts[6] = {75, 3, 162, 6, 648, 12};
        int off = 0;
        for (int si = 0; si < 6; ++si) {
            for (int i = t; i < cnts[si]; i += TPB) WB[off + i] = srcs[si][i];
            off += cnts[si];
        }
    }
    __syncthreads();
    const float* w1 = WB;        const float* b1 = WB + 75;
    const float* w2 = WB + 78;   const float* b2 = WB + 240;
    const float* w3 = WB + 246;  const float* b3 = WB + 894;

    // ---- conv1: 1->3ch, 5x5, s2, p2, 28x28 -> 14x14, ReLU ----------------
    for (int idx = t; idx < SPB * 3 * 196; idx += TPB) {
        const int s = idx / 588, rem = idx % 588;
        const int c = rem / 196, ij = rem % 196;
        const int i = ij / 14, j = ij % 14;
        const float* xs = x + (size_t)(s0 + s) * 784;
        float acc = b1[c];
#pragma unroll
        for (int u = 0; u < 5; ++u) {
            const int yy = 2 * i + u - 2;
            if (yy < 0 || yy >= 28) continue;
#pragma unroll
            for (int v = 0; v < 5; ++v) {
                const int xx = 2 * j + v - 2;
                if (xx < 0 || xx >= 28) continue;
                acc += w1[c * 25 + u * 5 + v] * xs[yy * 28 + xx];
            }
        }
        R1h[idx] = (_Float16)fmaxf(acc, 0.0f);
    }
    __syncthreads();

    // ---- conv2: 3->6ch, 3x3, s2, p1, 14x14 -> 7x7, ReLU ------------------
    for (int idx = t; idx < SPB * 6 * 49; idx += TPB) {
        const int s = idx / 294, rem = idx % 294;
        const int co = rem / 49, ij = rem % 49;
        const int i = ij / 7, j = ij % 7;
        float acc = b2[co];
        for (int ci = 0; ci < 3; ++ci)
#pragma unroll
            for (int u = 0; u < 3; ++u) {
                const int yy = 2 * i + u - 1;
                if (yy < 0 || yy >= 14) continue;
#pragma unroll
                for (int v = 0; v < 3; ++v) {
                    const int xx = 2 * j + v - 1;
                    if (xx < 0 || xx >= 14) continue;
                    acc += w2[(co * 3 + ci) * 9 + u * 3 + v] *
                           (float)R1h[s * 588 + ci * 196 + yy * 14 + xx];
                }
            }
        R2h[idx] = (_Float16)fmaxf(acc, 0.0f);
    }
    __syncthreads();

    // ---- conv3: 6->12ch, 3x3, s1, p1, 7x7 -> 7x7, ReLU (into R1) ---------
    for (int idx = t; idx < SPB * 12 * 49; idx += TPB) {
        const int s = idx / 588, rem = idx % 588;
        const int co = rem / 49, ij = rem % 49;
        const int i = ij / 7, j = ij % 7;
        float acc = b3[co];
        for (int ci = 0; ci < 6; ++ci)
#pragma unroll
            for (int u = 0; u < 3; ++u) {
                const int yy = i + u - 1;
                if (yy < 0 || yy >= 7) continue;
#pragma unroll
                for (int v = 0; v < 3; ++v) {
                    const int xx = j + v - 1;
                    if (xx < 0 || xx >= 7) continue;
                    acc += w3[(co * 6 + ci) * 9 + u * 3 + v] *
                           (float)R2h[s * 294 + ci * 49 + yy * 7 + xx];
                }
            }
        R1h[idx] = (_Float16)fmaxf(acc, 0.0f);
    }
    __syncthreads();

    // ---- adaptive pool 7x7 -> 2x2 (bins [0,4),[3,7)) ---------------------
    for (int idx = t; idx < SPB * 48; idx += TPB) {
        const int s = idx / 48, rem = idx % 48;
        const int c = rem / 4;
        const int rb = (rem >> 1) & 1, cb = rem & 1;
        const int r0 = rb ? 3 : 0, q0 = cb ? 3 : 0;
        float acc = 0.0f;
#pragma unroll
        for (int u = 0; u < 4; ++u)
#pragma unroll
            for (int v = 0; v < 4; ++v)
                acc += (float)R1h[s * 588 + c * 49 + (r0 + u) * 7 + (q0 + v)];
        R2f[idx] = acc * (1.0f / 16.0f);
    }
    __syncthreads();

    // ---- fc: 48 -> 256, ReLU (WMMA) --------------------------------------
    gemm_bias_relu<48, 256>(R2f, 48, fcp, fcb, R1f, 256, lane, m0);
    __syncthreads();

    // ---- kick TDM: packed U -> LDS (overlaps with normalization) ---------
#if HAVE_TDM
    if (t < 32) tdm_load_upk(Upk, OFF_R2);
#endif

    // ---- L2-normalize feat + init qout -----------------------------------
    if (t < SPB) {
        float* f = R1f + t * 256;
        float ss = 0.0f;
        for (int k = 0; k < 256; ++k) ss += f[k] * f[k];
        const float inv = 1.0f / fmaxf(sqrtf(ss), 1e-12f);
        for (int k = 0; k < 256; ++k) f[k] *= inv;
        qsh[t] = 0.0f;
    }

#if HAVE_TDM
    if (t < 32) __builtin_amdgcn_s_wait_tensorcnt(0);
#else
    {   // fallback: cooperative vector copy of packed U into LDS
        const float4* src = (const float4*)Upk;        // 8192 x 16B
        float4* dst = (float4*)(smem + OFF_R2);
        for (int i = t; i < 8192; i += TPB) dst[i] = src[i];
    }
#endif
    __syncthreads();

    // ---- quantum circuit as GEMM + signed |.|^2 (WMMA, B from LDS) -------
    qsim_accum(R1f, Ul, qsh, lane, m0);
    __syncthreads();

    // ---- cls1: scalar qout -> 180, ReLU ----------------------------------
    for (int idx = t; idx < SPB * 180; idx += TPB) {
        const int s = idx / 180, j = idx % 180;
        R1f[s * 180 + j] = fmaxf(qsh[s] * w1c[j] + b1c[j], 0.0f);
    }
    __syncthreads();

    // ---- classifier chain (WMMA) -----------------------------------------
    gemm_bias_relu<180, 140>(R1f, 180, c2p, b2c, R2f, 140, lane, m0);
    __syncthreads();
    gemm_bias_relu<140, 100>(R2f, 140, c3p, b3c, R3f, 100, lane, m0);
    __syncthreads();
    gemm_bias_relu<100, 50>(R3f, 100, c4p, b4c, R1f, 50, lane, m0);
    __syncthreads();

    // ---- cls5: 50 -> 1, sigmoid ------------------------------------------
    if (t < SPB) {
        const float* h = R1f + t * 50;
        float z = b5c[0];
        for (int k = 0; k < 50; ++k) z += h[k] * w5c[k];
        out[s0 + t] = 1.0f / (1.0f + expf(-z));
    }
}

// ===========================================================================
extern "C" void kernel_launch(void* const* d_in, const int* in_sizes, int n_in,
                              void* d_out, int out_size, void* d_ws, size_t ws_size,
                              hipStream_t stream) {
    const float* x   = (const float*)d_in[0];
    const float* c1w = (const float*)d_in[1];
    const float* c1b = (const float*)d_in[2];
    const float* c2w = (const float*)d_in[3];
    const float* c2b = (const float*)d_in[4];
    const float* c3w = (const float*)d_in[5];
    const float* c3b = (const float*)d_in[6];
    const float* fcw = (const float*)d_in[7];
    const float* fcb = (const float*)d_in[8];
    const float* qw  = (const float*)d_in[9];
    const float* w1c = (const float*)d_in[10];
    const float* b1c = (const float*)d_in[11];
    const float* w2c = (const float*)d_in[12];
    const float* b2c = (const float*)d_in[13];
    const float* w3c = (const float*)d_in[14];
    const float* b3c = (const float*)d_in[15];
    const float* w4c = (const float*)d_in[16];
    const float* b4c = (const float*)d_in[17];
    const float* w5c = (const float*)d_in[18];
    const float* b5c = (const float*)d_in[19];
    float* out = (float*)d_out;

    _Float16* wsh = (_Float16*)d_ws;
    _Float16* Upk = wsh + WS_UPK;
    _Float16* fcp = wsh + WS_FCP;
    _Float16* c2p = wsh + WS_C2P;
    _Float16* c3p = wsh + WS_C3P;
    _Float16* c4p = wsh + WS_C4P;

    const int B = in_sizes[0] / 784;           // 65536

    build_u_kernel<<<1, TPB, QDIM * QDIM * sizeof(float), stream>>>(qw, Upk);
    pack_w_kernel<<<(16 * 2 * 512 + 255) / 256, 256, 0, stream>>>(fcw, 256, 48, fcp);
    pack_w_kernel<<<(9 * 6 * 512 + 255) / 256, 256, 0, stream>>>(w2c, 140, 180, c2p);
    pack_w_kernel<<<(7 * 5 * 512 + 255) / 256, 256, 0, stream>>>(w3c, 100, 140, c3p);
    pack_w_kernel<<<(4 * 4 * 512 + 255) / 256, 256, 0, stream>>>(w4c, 50, 100, c4p);

    fused_kernel<<<B / SPB, TPB, SMEM_BYTES, stream>>>(
        x, c1w, c1b, c2w, c2b, c3w, c3b, fcp, fcb, Upk,
        w1c, b1c, c2p, b2c, c3p, b3c, c4p, b4c, w5c, b5c, out);
}
